// TritonDynamicLinear_60155311948051
// MI455X (gfx1250) — matrix operand order repaired and checked by reading, not compiled, not run on hardware
//
#include <hip/hip_runtime.h>

// Problem constants (from reference): B=2, S=2048, K=4096, N=4096; 32x32 blocks.
#define MDIM 4096   // B*S
#define KDIM 4096
#define NDIM 4096
#define BLK  32
#define NB   (NDIM / BLK)   // 128 N-blocks
#define KB   (KDIM / BLK)   // 128 K-blocks
#define BM   128            // workgroup M tile
#define BN   32             // workgroup N tile (== one N-block)
#define LDA  40             // padded LDS row stride (bf16): 80B -> conflict-free b128

typedef __bf16 bf16_t;
typedef __attribute__((ext_vector_type(16))) __bf16 v16bf;
typedef __attribute__((ext_vector_type(8)))  __bf16 v8bf;
typedef __attribute__((ext_vector_type(4)))  __bf16 v4bf;
typedef __attribute__((ext_vector_type(8)))  float  v8f;
typedef __attribute__((__vector_size__(4 * sizeof(int)))) int v4i;

#define AS1 __attribute__((address_space(1)))
#define AS3 __attribute__((address_space(3)))

union FragAB { v16bf v; v8bf h[2]; };

// ---- CDNA5 async copy: global -> LDS (b128), tracked by ASYNCcnt ----------
__device__ __forceinline__ void async_copy_b128(void* lds, const void* gptr) {
#if __has_builtin(__builtin_amdgcn_global_load_async_to_lds_b128)
  __builtin_amdgcn_global_load_async_to_lds_b128(
      (AS1 v4i*)gptr, (AS3 v4i*)lds, 0, 0);
#else
  unsigned loff = (unsigned)(uintptr_t)(AS3 char*)lds;        // LDS byte offset
  unsigned long long ga = (unsigned long long)(uintptr_t)gptr;
  asm volatile("global_load_async_to_lds_b128 %0, %1, off" :: "v"(loff), "v"(ga)
               : "memory");
#endif
}

template <int N>
__device__ __forceinline__ void wait_asynccnt() {
#if __has_builtin(__builtin_amdgcn_s_wait_asynccnt)
  __builtin_amdgcn_s_wait_asynccnt(N);
#else
  asm volatile("s_wait_asynccnt %0" :: "i"(N));
#endif
}

// ---------------- Kernel 1: 32x32 block-mask pooling (any-nonzero) ----------
__global__ __launch_bounds__(256) void block_mask_kernel(const int* __restrict__ mask,
                                                         int* __restrict__ bm) {
  const int wid  = (blockIdx.x * 256 + threadIdx.x) >> 5;  // one wave per block tile
  const int lane = threadIdx.x & 31;
  const int nb = wid / KB;
  const int kb = wid % KB;
  const int4* row = reinterpret_cast<const int4*>(
      mask + (size_t)(nb * BLK + lane) * KDIM + kb * BLK);
  int acc = 0;
#pragma unroll
  for (int j = 0; j < 8; ++j) {
    int4 v = row[j];
    acc |= v.x | v.y | v.z | v.w;
  }
  unsigned long long anyb = __ballot(acc != 0);
  if (lane == 0) bm[wid] = (anyb != 0ull) ? 1 : 0;
}

// ---------------- Kernel 2: compact active kb list per nb -------------------
__global__ void compact_kernel(const int* __restrict__ bm,
                               int* __restrict__ counts,
                               int* __restrict__ lists) {
  const int nb = blockIdx.x;
  if (threadIdx.x == 0) {
    int c = 0;
    for (int kb = 0; kb < KB; ++kb)
      if (bm[nb * KB + kb]) lists[nb * KB + c++] = kb;
    counts[nb] = c;
  }
}

// ---------------- Kernel 3a: one-time f32 -> (hi,lo) bf16 split -------------
__global__ __launch_bounds__(256) void split_convert_kernel(const float* __restrict__ src,
                                                            bf16_t* __restrict__ hi,
                                                            bf16_t* __restrict__ lo) {
  const size_t i = ((size_t)blockIdx.x * 256 + threadIdx.x) * 8;
  float4 a = *reinterpret_cast<const float4*>(src + i);
  float4 b = *reinterpret_cast<const float4*>(src + i + 4);
  float xs[8] = {a.x, a.y, a.z, a.w, b.x, b.y, b.z, b.w};
  v8bf h, l;
#pragma unroll
  for (int e = 0; e < 8; ++e) {
    __bf16 hb = (__bf16)xs[e];
    h[e] = hb;
    l[e] = (__bf16)(xs[e] - (float)hb);
  }
  *reinterpret_cast<v8bf*>(hi + i) = h;
  *reinterpret_cast<v8bf*>(lo + i) = l;
}

// ------- Kernel 3b: block-sparse GEMM, async-LDS double-buffered ------------
__global__ __launch_bounds__(256) void bsgemm_async_kernel(
    const bf16_t* __restrict__ Dhi, const bf16_t* __restrict__ Dlo,
    const bf16_t* __restrict__ Whig, const bf16_t* __restrict__ Wlog,
    const float* __restrict__ bias, const int* __restrict__ counts,
    const int* __restrict__ lists, float* __restrict__ out) {
  __shared__ __align__(16) bf16_t Ah[2][BM][LDA];
  __shared__ __align__(16) bf16_t Al[2][BM][LDA];
  __shared__ __align__(16) bf16_t Wh[2][BN][LDA];
  __shared__ __align__(16) bf16_t Wl[2][BN][LDA];

  const int tid  = threadIdx.x;
  const int wave = tid >> 5;
  const int lane = tid & 31;
  const int m0 = blockIdx.x * BM;
  const int nb = blockIdx.y;
  const int n0 = nb * BN;

  v8f c0 = {};
  v8f c1 = {};
  const int  cnt = counts[nb];
  const int* lst = lists + nb * KB;

  // Per-thread async-copy coordinates (5 b128 copies per thread per tile set).
  const int ar0 = tid >> 2, ac0 = (tid & 3) * 8;          // A rows 0..63
  const int ar1 = ar0 + 64;                               // A rows 64..127
  const int wsel = tid >> 7;                              // 0 -> Wh, 1 -> Wl
  const int wc   = tid & 127;
  const int wr   = wc >> 2, wcol = (wc & 3) * 8;

  // Fragment coordinates (contiguous k = 16*halfSel + s for both A and B).
  const int halfSel = lane >> 4;
  const int lane15  = lane & 15;
  const int mrow  = wave * 16 + lane15;
  const int akoff = halfSel * 16;
  const int bkoff = halfSel * 16;

  auto issue_set = [&](int set, int k0) {
    const int b = set & 1;
    async_copy_b128(&Ah[b][ar0][ac0], Dhi + (size_t)(m0 + ar0) * KDIM + k0 + ac0);
    async_copy_b128(&Ah[b][ar1][ac0], Dhi + (size_t)(m0 + ar1) * KDIM + k0 + ac0);
    async_copy_b128(&Al[b][ar0][ac0], Dlo + (size_t)(m0 + ar0) * KDIM + k0 + ac0);
    async_copy_b128(&Al[b][ar1][ac0], Dlo + (size_t)(m0 + ar1) * KDIM + k0 + ac0);
    const bf16_t* wsrc = wsel ? (Wlog + (size_t)(n0 + wr) * KDIM + k0 + wcol)
                              : (Whig + (size_t)(n0 + wr) * KDIM + k0 + wcol);
    void* wdst = wsel ? (void*)&Wl[b][wr][wcol] : (void*)&Wh[b][wr][wcol];
    async_copy_b128(wdst, wsrc);
  };

  if (cnt > 0) issue_set(0, lst[0] * BLK);

  for (int it = 0; it < cnt; ++it) {
    const int cur = it & 1;
    if (it + 1 < cnt) {
      issue_set(it + 1, lst[it + 1] * BLK);
      wait_asynccnt<5>();   // set `it` (first 5) landed; set `it+1` in flight
    } else {
      wait_asynccnt<0>();
    }
    __syncthreads();        // publish tile `it` to all waves

    FragAB ah, al, bh0, bh1, bl0, bl1;
    ah.h[0] = *reinterpret_cast<const v8bf*>(&Ah[cur][mrow][akoff]);
    ah.h[1] = *reinterpret_cast<const v8bf*>(&Ah[cur][mrow][akoff + 8]);
    al.h[0] = *reinterpret_cast<const v8bf*>(&Al[cur][mrow][akoff]);
    al.h[1] = *reinterpret_cast<const v8bf*>(&Al[cur][mrow][akoff + 8]);

    bh0.h[0] = *reinterpret_cast<const v8bf*>(&Wh[cur][lane15][bkoff]);
    bh0.h[1] = *reinterpret_cast<const v8bf*>(&Wh[cur][lane15][bkoff + 8]);
    bl0.h[0] = *reinterpret_cast<const v8bf*>(&Wl[cur][lane15][bkoff]);
    bl0.h[1] = *reinterpret_cast<const v8bf*>(&Wl[cur][lane15][bkoff + 8]);
    bh1.h[0] = *reinterpret_cast<const v8bf*>(&Wh[cur][16 + lane15][bkoff]);
    bh1.h[1] = *reinterpret_cast<const v8bf*>(&Wh[cur][16 + lane15][bkoff + 8]);
    bl1.h[0] = *reinterpret_cast<const v8bf*>(&Wl[cur][16 + lane15][bkoff]);
    bl1.h[1] = *reinterpret_cast<const v8bf*>(&Wl[cur][16 + lane15][bkoff + 8]);

    // Split-bf16 products: hi*hi + hi*lo + lo*hi.
    c0 = __builtin_amdgcn_wmma_f32_16x16x32_bf16(false, ah.v, false, bh0.v, (short)0, c0, false, false);
    c0 = __builtin_amdgcn_wmma_f32_16x16x32_bf16(false, ah.v, false, bl0.v, (short)0, c0, false, false);
    c0 = __builtin_amdgcn_wmma_f32_16x16x32_bf16(false, al.v, false, bh0.v, (short)0, c0, false, false);
    c1 = __builtin_amdgcn_wmma_f32_16x16x32_bf16(false, ah.v, false, bh1.v, (short)0, c1, false, false);
    c1 = __builtin_amdgcn_wmma_f32_16x16x32_bf16(false, ah.v, false, bl1.v, (short)0, c1, false, false);
    c1 = __builtin_amdgcn_wmma_f32_16x16x32_bf16(false, al.v, false, bh1.v, (short)0, c1, false, false);

    __syncthreads();        // all reads of buf[cur] done before it is refilled
  }

  // Epilogue: C layout M = v + 8*halfSel, N = lane15.
  const int moff = halfSel * 8;
  const float b0 = bias[n0 + lane15];
  const float b1 = bias[n0 + 16 + lane15];
#pragma unroll
  for (int v = 0; v < 8; ++v) {
    const size_t r = (size_t)(m0 + wave * 16 + moff + v) * NDIM;
    out[r + n0 + lane15]      = c0[v] + b0;
    out[r + n0 + 16 + lane15] = c1[v] + b1;
  }
}

// ------- Kernel 3c: fused fallback (used only if ws too small) --------------
__global__ __launch_bounds__(256) void bsgemm_kernel(const float* __restrict__ data,
                                                     const float* __restrict__ weight,
                                                     const float* __restrict__ bias,
                                                     const int* __restrict__ counts,
                                                     const int* __restrict__ lists,
                                                     float* __restrict__ out) {
  __shared__ __align__(16) bf16_t Ah[BM][LDA];
  __shared__ __align__(16) bf16_t Al[BM][LDA];
  __shared__ __align__(16) bf16_t Wh[BN][LDA];
  __shared__ __align__(16) bf16_t Wl[BN][LDA];

  const int tid  = threadIdx.x;
  const int wave = tid >> 5;
  const int lane = tid & 31;
  const int m0 = blockIdx.x * BM;
  const int nb = blockIdx.y;
  const int n0 = nb * BN;

  v8f c0 = {};
  v8f c1 = {};
  const int  cnt = counts[nb];
  const int* lst = lists + nb * KB;

  const int halfSel = lane >> 4;
  const int lane15  = lane & 15;
  const int mrow  = wave * 16 + lane15;
  const int akoff = halfSel * 16;
  const int bkoff = halfSel * 16;

  for (int it = 0; it < cnt; ++it) {
    const int k0 = lst[it] * BLK;
    __syncthreads();
#pragma unroll
    for (int j = 0; j < 4; ++j) {
      const int f = tid + j * 256;
      const int row = f >> 3;
      const int col = (f & 7) * 4;
      float4 v = *reinterpret_cast<const float4*>(
          data + (size_t)(m0 + row) * KDIM + k0 + col);
      float xs[4] = {v.x, v.y, v.z, v.w};
      v4bf h, l;
#pragma unroll
      for (int e = 0; e < 4; ++e) {
        __bf16 hb = (__bf16)xs[e];
        h[e] = hb;
        l[e] = (__bf16)(xs[e] - (float)hb);
      }
      *reinterpret_cast<v4bf*>(&Ah[row][col]) = h;
      *reinterpret_cast<v4bf*>(&Al[row][col]) = l;
    }
    {
      const int row = tid >> 3;
      const int col = (tid & 7) * 4;
      float4 v = *reinterpret_cast<const float4*>(
          weight + (size_t)(n0 + row) * KDIM + k0 + col);
      float xs[4] = {v.x, v.y, v.z, v.w};
      v4bf h, l;
#pragma unroll
      for (int e = 0; e < 4; ++e) {
        __bf16 hb = (__bf16)xs[e];
        h[e] = hb;
        l[e] = (__bf16)(xs[e] - (float)hb);
      }
      *reinterpret_cast<v4bf*>(&Wh[row][col]) = h;
      *reinterpret_cast<v4bf*>(&Wl[row][col]) = l;
    }
    __syncthreads();

    FragAB ah, al, bh0, bh1, bl0, bl1;
    ah.h[0] = *reinterpret_cast<const v8bf*>(&Ah[mrow][akoff]);
    ah.h[1] = *reinterpret_cast<const v8bf*>(&Ah[mrow][akoff + 8]);
    al.h[0] = *reinterpret_cast<const v8bf*>(&Al[mrow][akoff]);
    al.h[1] = *reinterpret_cast<const v8bf*>(&Al[mrow][akoff + 8]);
    bh0.h[0] = *reinterpret_cast<const v8bf*>(&Wh[lane15][bkoff]);
    bh0.h[1] = *reinterpret_cast<const v8bf*>(&Wh[lane15][bkoff + 8]);
    bl0.h[0] = *reinterpret_cast<const v8bf*>(&Wl[lane15][bkoff]);
    bl0.h[1] = *reinterpret_cast<const v8bf*>(&Wl[lane15][bkoff + 8]);
    bh1.h[0] = *reinterpret_cast<const v8bf*>(&Wh[16 + lane15][bkoff]);
    bh1.h[1] = *reinterpret_cast<const v8bf*>(&Wh[16 + lane15][bkoff + 8]);
    bl1.h[0] = *reinterpret_cast<const v8bf*>(&Wl[16 + lane15][bkoff]);
    bl1.h[1] = *reinterpret_cast<const v8bf*>(&Wl[16 + lane15][bkoff + 8]);

    c0 = __builtin_amdgcn_wmma_f32_16x16x32_bf16(false, ah.v, false, bh0.v, (short)0, c0, false, false);
    c0 = __builtin_amdgcn_wmma_f32_16x16x32_bf16(false, ah.v, false, bl0.v, (short)0, c0, false, false);
    c0 = __builtin_amdgcn_wmma_f32_16x16x32_bf16(false, al.v, false, bh0.v, (short)0, c0, false, false);
    c1 = __builtin_amdgcn_wmma_f32_16x16x32_bf16(false, ah.v, false, bh1.v, (short)0, c1, false, false);
    c1 = __builtin_amdgcn_wmma_f32_16x16x32_bf16(false, ah.v, false, bl1.v, (short)0, c1, false, false);
    c1 = __builtin_amdgcn_wmma_f32_16x16x32_bf16(false, al.v, false, bh1.v, (short)0, c1, false, false);
  }

  const int moff = halfSel * 8;
  const float b0 = bias[n0 + lane15];
  const float b1 = bias[n0 + 16 + lane15];
#pragma unroll
  for (int v = 0; v < 8; ++v) {
    const size_t r = (size_t)(m0 + wave * 16 + moff + v) * NDIM;
    out[r + n0 + lane15]      = c0[v] + b0;
    out[r + n0 + 16 + lane15] = c1[v] + b1;
  }
}

// ---------------------------------------------------------------------------
extern "C" void kernel_launch(void* const* d_in, const int* in_sizes, int n_in,
                              void* d_out, int out_size, void* d_ws, size_t ws_size,
                              hipStream_t stream) {
  (void)in_sizes; (void)n_in; (void)out_size;
  const float* data   = (const float*)d_in[0];
  const int*   mask   = (const int*)d_in[1];
  const float* weight = (const float*)d_in[2];
  const float* bias   = (const float*)d_in[3];
  float* out = (float*)d_out;

  int* bm     = (int*)d_ws;          // NB*KB ints
  int* counts = bm + NB * KB;        // NB ints
  int* lists  = counts + NB;         // NB*KB ints

  block_mask_kernel<<<(NB * KB) / 8, 256, 0, stream>>>(mask, bm);
  compact_kernel<<<NB, 64, 0, stream>>>(bm, counts, lists);

  const size_t elems    = (size_t)MDIM * KDIM;      // == N*K as well
  const size_t conv_off = 256 * 1024;
  const size_t need     = conv_off + 4 * elems * sizeof(bf16_t);
  dim3 grid(MDIM / BM, NB);

  if (ws_size >= need) {
    bf16_t* Dhi = (bf16_t*)((char*)d_ws + conv_off);
    bf16_t* Dlo = Dhi + elems;
    bf16_t* Whi = Dlo + elems;
    bf16_t* Wlo = Whi + elems;
    const int cgrid = (int)(elems / (8 * 256));     // 8192
    split_convert_kernel<<<cgrid, 256, 0, stream>>>(data, Dhi, Dlo);
    split_convert_kernel<<<cgrid, 256, 0, stream>>>(weight, Whi, Wlo);
    bsgemm_async_kernel<<<grid, 256, 0, stream>>>(Dhi, Dlo, Whi, Wlo, bias,
                                                  counts, lists, out);
  } else {
    bsgemm_kernel<<<grid, 256, 0, stream>>>(data, weight, bias, counts, lists, out);
  }
}
